// MPGRUImputer_61933428408502
// MI455X (gfx1250) — compile-verified, matrix operand order checked by reading, not run on hardware
//
#include <hip/hip_runtime.h>
#include <hip/hip_bf16.h>

// ---------------- problem constants ----------------
#define NN   1024   // nodes
#define BB   8      // batch
#define HH   64     // hidden
#define TT   128    // timesteps
#define C1   66     // channels per diffusion part (2F + H)
#define CP   352    // padded channels per batch (5*66=330 -> 352 = 11*32)

typedef __attribute__((ext_vector_type(16))) __bf16 v16bf;
typedef __attribute__((ext_vector_type(8)))  __bf16 v8bf;
typedef __attribute__((ext_vector_type(8)))  float  v8f;

#define WMMA_BF16(a,b,c) \
  __builtin_amdgcn_wmma_f32_16x16x32_bf16(false,(a),false,(b),(short)0,(c),false,false)

// ---- CDNA5 async global->LDS staging (ASYNCcnt-tracked) via inline asm ----
// GLOBAL_LOAD_ASYNC_TO_LDS_B128: VDST = LDS byte address VGPR, VADDR = 64-bit
// global address pair, no SADDR ("off"). Tracked by ASYNCcnt.
#define USE_ASYNC 1

#if USE_ASYNC
static __device__ __forceinline__ void async_cp16(const __bf16* g, __bf16* l) {
  const unsigned lds = (unsigned)(uintptr_t)l;  // flat LDS addr[31:0] == offset
  const unsigned long long ga = (unsigned long long)(uintptr_t)g;
  asm volatile("global_load_async_to_lds_b128 %0, %1, off"
               :
               : "v"(lds), "v"(ga)
               : "memory");
}
static __device__ __forceinline__ void async_wait() {
  asm volatile("s_wait_asynccnt 0x0" ::: "memory");
}
#endif

// bf16 16x16x32 WMMA operand fragment: lane L (idx = L&15, half = L>>4) holds
// K in [8*half, 8*half+8) then [16+8*half, 16+8*half+8)  -> two b128 LDS loads
// from a K-contiguous row.
static __device__ __forceinline__ v16bf load_frag(const __bf16* row, int hf) {
  v8bf lo = *(const v8bf*)(row + 8 * hf);
  v8bf hi = *(const v8bf*)(row + 16 + 8 * hf);
  v16bf f;
#pragma unroll
  for (int i = 0; i < 8; ++i) { f[i] = lo[i]; f[i + 8] = hi[i]; }
  return f;
}

// ---------------- setup kernels ----------------
__global__ __launch_bounds__(256) void k_sums(const float* __restrict__ adj,
                                              float* __restrict__ invR,
                                              float* __restrict__ invC) {
  const int j = blockIdx.x, t = threadIdx.x;
  float rs = 0.f, cs = 0.f;
  for (int v = t; v < NN; v += 256) {
    rs += adj[(size_t)j * NN + v];
    cs += adj[(size_t)v * NN + j];
  }
  __shared__ float sR[256], sC[256];
  sR[t] = rs; sC[t] = cs; __syncthreads();
  for (int s = 128; s > 0; s >>= 1) {
    if (t < s) { sR[t] += sR[t + s]; sC[t] += sC[t + s]; }
    __syncthreads();
  }
  if (t == 0) {
    invR[j] = 1.f / (sR[0] + 1e-8f);
    invC[j] = 1.f / (sC[0] + 1e-8f);
  }
}

// supF[w][v] = adj[w][v]/rowsum(adj)[w]; supB[w][v] = adj[v][w]/colsum(adj)[w]
// These are exactly the "Bt[n][k]" layouts the WMMA B fragment wants.
__global__ __launch_bounds__(256) void k_buildsup(const float* __restrict__ adj,
                                                  const float* __restrict__ invR,
                                                  const float* __restrict__ invC,
                                                  __bf16* __restrict__ supF,
                                                  __bf16* __restrict__ supB) {
  const int idx = blockIdx.x * 256 + threadIdx.x;  // w*NN + v
  const int w = idx >> 10, v = idx & (NN - 1);
  supF[idx] = (__bf16)(adj[idx] * invR[w]);
  supB[idx] = (__bf16)(adj[(size_t)v * NN + w] * invC[w]);
}

__global__ __launch_bounds__(256) void k_buildw(const float* __restrict__ Wr,
                                                const float* __restrict__ Wu,
                                                const float* __restrict__ Wc,
                                                const float* __restrict__ br,
                                                const float* __restrict__ bu,
                                                __bf16* __restrict__ Wru,
                                                __bf16* __restrict__ Wcb,
                                                float* __restrict__ brbu) {
  const int idx = blockIdx.x * 256 + threadIdx.x;
  if (idx < 128 * CP) {
    const int o = idx / CP, k = idx - o * CP;
    float v = 0.f;
    if (k < 330) v = (o < 64) ? Wr[o * 330 + k] : Wu[(o - 64) * 330 + k];
    Wru[idx] = (__bf16)v;
  }
  if (idx < 64 * CP) {
    const int o = idx / CP, k = idx - o * CP;
    Wcb[idx] = (__bf16)((k < 330) ? Wc[o * 330 + k] : 0.f);
  }
  if (idx < 128) brbu[idx] = (idx < 64) ? br[idx] : bu[idx - 64];
}

__global__ __launch_bounds__(256) void k_zero(float* __restrict__ h,
                                              __bf16* __restrict__ zc) {
  const int idx = blockIdx.x * 256 + threadIdx.x;
  if (idx < BB * HH * NN) h[idx] = 0.f;
  if (idx < BB * CP * NN) zc[idx] = (__bf16)0.f;
}

// ---------------- per-step elementwise kernels ----------------
// x_hat = Wout.h + bout ; stage preds/states ; write zc part0 = [x_in, mask, h]
__global__ __launch_bounds__(256) void k_prep(const float* __restrict__ h,
                                              const float* __restrict__ x,
                                              const unsigned char* __restrict__ mask,
                                              const float* __restrict__ Wout,
                                              const float* __restrict__ bout,
                                              __bf16* __restrict__ zc,
                                              float* __restrict__ predsStage,
                                              float* __restrict__ statesStage,
                                              int t) {
  const int idx = blockIdx.x * 256 + threadIdx.x;  // 0..BB*NN-1
  const int b = idx >> 10, n = idx & (NN - 1);
  const float* hb = h + (size_t)(b * HH) * NN + n;
  float xh = bout[0];
#pragma unroll 8
  for (int k = 0; k < HH; ++k) xh += Wout[k] * hb[(size_t)k * NN];
  const int sl = t & 15;
  predsStage[sl * (BB * NN) + idx] = xh;
  const size_t xi = ((size_t)(b * NN) + n) * TT + t;
  const unsigned char m = mask[xi];
  const float xin = m ? x[xi] : xh;
  __bf16* z0 = zc + (size_t)(b * CP) * NN + n;
  z0[0] = (__bf16)xin;
  z0[NN] = (__bf16)(m ? 1.f : 0.f);
  float* ss = statesStage + (size_t)sl * (BB * HH * NN) + (size_t)(b * HH) * NN + n;
#pragma unroll 8
  for (int k = 0; k < HH; ++k) {
    const float hv = hb[(size_t)k * NN];
    z0[(size_t)(2 + k) * NN] = (__bf16)hv;
    ss[(size_t)k * NN] = hv;
  }
}

// r = sigmoid(ru[0:64]); u = sigmoid(ru[64:128]) stored in place; z0[2..] = r*h
__global__ __launch_bounds__(256) void k_gate(float* __restrict__ ru,
                                              const float* __restrict__ h,
                                              __bf16* __restrict__ zc) {
  const int idx = blockIdx.x * 256 + threadIdx.x;  // (b*64+hc)*NN + n
  const int n = idx & (NN - 1), hc = (idx >> 10) & 63, b = idx >> 16;
  const size_t ri = ((size_t)(b * 128) + hc) * NN + n;
  const size_t ui = ((size_t)(b * 128) + 64 + hc) * NN + n;
  const float r = 1.f / (1.f + __expf(-ru[ri]));
  const float u = 1.f / (1.f + __expf(-ru[ui]));
  ru[ui] = u;
  zc[((size_t)(b * CP) + 2 + hc) * NN + n] = (__bf16)(r * h[idx]);
}

// h = u*h + (1-u)*tanh(c)
__global__ __launch_bounds__(256) void k_update(const float* __restrict__ ru,
                                                const float* __restrict__ cg,
                                                float* __restrict__ h) {
  const int idx = blockIdx.x * 256 + threadIdx.x;
  const int n = idx & (NN - 1), hc = (idx >> 10) & 63, b = idx >> 16;
  const float u = ru[((size_t)(b * 128) + 64 + hc) * NN + n];
  const float c = tanhf(cg[idx]);
  h[idx] = u * h[idx] + (1.f - u) * c;
}

// ---------------- staged-output flush (contiguous 64B stores) ----------------
__global__ __launch_bounds__(256) void k_flush_states(const float* __restrict__ stage,
                                                      float* __restrict__ out, int t0) {
  const int idx = blockIdx.x * 256 + threadIdx.x;  // (b*64+hh)*NN + n
  float v[16];
#pragma unroll
  for (int j = 0; j < 16; ++j) v[j] = stage[(size_t)j * (BB * HH * NN) + idx];
  float4* dst = (float4*)(out + (size_t)idx * TT + t0);
#pragma unroll
  for (int j = 0; j < 4; ++j)
    dst[j] = make_float4(v[4 * j], v[4 * j + 1], v[4 * j + 2], v[4 * j + 3]);
}

__global__ __launch_bounds__(256) void k_flush_preds(const float* __restrict__ stage,
                                                     float* __restrict__ out, int t0) {
  const int idx = blockIdx.x * 256 + threadIdx.x;  // b*NN + n
  float v[16];
#pragma unroll
  for (int j = 0; j < 16; ++j) v[j] = stage[(size_t)j * (BB * NN) + idx];
  float4* dst = (float4*)(out + (size_t)idx * TT + t0);
#pragma unroll
  for (int j = 0; j < 4; ++j)
    dst[j] = make_float4(v[4 * j], v[4 * j + 1], v[4 * j + 2], v[4 * j + 3]);
}

// ---------------- diffusion GEMM:  zc[dst] = zc[src] x sup^T  ----------------
// out[r][w] = sum_v Z[r][v] * sup[w][v]; M=528 (row-mapped b*CP+off+c),
// N=1024, K=1024, bf16 in/out, f32 WMMA accumulate.
// Tile: BM=48, BN=128, BK=32, 8 waves; wave w owns N-subtile w (3 wmma/iter).
// gridDim.z = 2 merges the independent forward / backward support chains.
__global__ __launch_bounds__(256) void k_gemm_diff(__bf16* __restrict__ zc,
                                                   const __bf16* __restrict__ supF,
                                                   const __bf16* __restrict__ supB,
                                                   int srcF, int dstF,
                                                   int srcB, int dstB) {
  __shared__ __align__(16) __bf16 lA[2][48 * 40];
  __shared__ __align__(16) __bf16 lB[2][128 * 40];
  const int tid = threadIdx.x;
  const int n0 = blockIdx.x * 128;
  const int r0 = blockIdx.y * 48;
  const __bf16* sup = blockIdx.z ? supB : supF;
  const int srcOff  = blockIdx.z ? srcB : srcF;
  const int dstOff  = blockIdx.z ? dstB : dstF;

  // A staging piece: 48 rows x 4 chunks of 16B (tid<192)
  const __bf16* aPtr = nullptr; int aLds = 0;
  if (tid < 192) {
    const int r = r0 + (tid >> 2);
    const int b = r / C1, c = r - b * C1;
    aPtr = zc + (size_t)((b * CP) + srcOff + c) * NN + (tid & 3) * 8;
    aLds = (tid >> 2) * 40 + (tid & 3) * 8;
  }
  // B staging: 128 rows x 4 chunks -> 2 pieces/thread (sup rows are already Bt)
  const __bf16* bPtr0 = sup + (size_t)(n0 + (tid >> 2)) * NN + (tid & 3) * 8;
  const int bLds0 = (tid >> 2) * 40 + (tid & 3) * 8;
  const int p1 = tid + 256;
  const __bf16* bPtr1 = sup + (size_t)(n0 + (p1 >> 2)) * NN + (p1 & 3) * 8;
  const int bLds1 = (p1 >> 2) * 40 + (p1 & 3) * 8;

#if USE_ASYNC
  if (tid < 192) async_cp16(aPtr, &lA[0][aLds]);
  async_cp16(bPtr0, &lB[0][bLds0]);
  async_cp16(bPtr1, &lB[0][bLds1]);
  async_wait();
#else
  int4 ra = {0, 0, 0, 0}, rb0, rb1;
  if (tid < 192) ra = *(const int4*)aPtr;
  rb0 = *(const int4*)bPtr0;
  rb1 = *(const int4*)bPtr1;
  if (tid < 192) *(int4*)&lA[0][aLds] = ra;
  *(int4*)&lB[0][bLds0] = rb0;
  *(int4*)&lB[0][bLds1] = rb1;
#endif
  __syncthreads();

  const int lane = tid & 31, wv = tid >> 5;
  const int ml = lane & 15, hf = lane >> 4;
  v8f acc[3] = {};

  for (int kt = 0; kt < 32; ++kt) {
    const int cur = kt & 1, nxt = cur ^ 1;
    if (kt < 31) {
      const int ko = (kt + 1) * 32;
#if USE_ASYNC
      if (tid < 192) async_cp16(aPtr + ko, &lA[nxt][aLds]);
      async_cp16(bPtr0 + ko, &lB[nxt][bLds0]);
      async_cp16(bPtr1 + ko, &lB[nxt][bLds1]);
#else
      if (tid < 192) ra = *(const int4*)(aPtr + ko);
      rb0 = *(const int4*)(bPtr0 + ko);
      rb1 = *(const int4*)(bPtr1 + ko);
#endif
    }
    const __bf16* Ab = &lA[cur][0];
    const __bf16* Bb = &lB[cur][0];
    const v16bf bf = load_frag(Bb + (wv * 16 + ml) * 40, hf);
#pragma unroll
    for (int i = 0; i < 3; ++i) {
      const v16bf af = load_frag(Ab + (i * 16 + ml) * 40, hf);
      acc[i] = WMMA_BF16(af, bf, acc[i]);
    }
    if (kt < 31) {
#if USE_ASYNC
      async_wait();
#else
      if (tid < 192) *(int4*)&lA[nxt][aLds] = ra;
      *(int4*)&lB[nxt][bLds0] = rb0;
      *(int4*)&lB[nxt][bLds1] = rb1;
#endif
    }
    __syncthreads();
  }

  // C layout: VGPR r -> M = r + (lane>=16 ? 8 : 0), N = n0 + wv*16 + (lane&15)
  const int n = n0 + wv * 16 + ml;
#pragma unroll
  for (int i = 0; i < 3; ++i) {
#pragma unroll
    for (int r = 0; r < 8; ++r) {
      const int m = r0 + i * 16 + hf * 8 + r;
      const int b = m / C1, c = m - b * C1;
      zc[(size_t)((b * CP) + dstOff + c) * NN + n] = (__bf16)acc[i][r];
    }
  }
}

// ---------------- projection GEMM (per batch):  out = W x zc_b + bias -------
// A = W [M, CP] row-major bf16 ; B = zc_b [CP, NN] (transposed while staging)
// Tile: BM=64, BN=128, BK=32, KT = CP/32 = 11, f32 output.
__global__ __launch_bounds__(256) void k_gemm_proj(const __bf16* __restrict__ W,
                                                   const __bf16* __restrict__ zc,
                                                   float* __restrict__ out,
                                                   const float* __restrict__ bias,
                                                   int M) {
  __shared__ __align__(16) __bf16 lA[2][64 * 40];
  __shared__ __align__(16) __bf16 lB[2][128 * 40];
  const int tid = threadIdx.x;
  const int n0 = blockIdx.x * 128;
  const int m0 = blockIdx.y * 64;
  const int b  = blockIdx.z;
  const __bf16* zb = zc + (size_t)b * CP * NN;

  const __bf16* aPtr = W + (size_t)(m0 + (tid >> 2)) * CP + (tid & 3) * 8;
  const int aLds = (tid >> 2) * 40 + (tid & 3) * 8;

  // B transpose staging: thread = (kpair, nchunk); loads rows 2kp,2kp+1 and
  // packs {lo,hi} 16-bit pairs into b32 LDS stores at Bt[n][2kp].
  const int kp = tid >> 4, nch = tid & 15;
  const __bf16* bPtrA = zb + (size_t)(2 * kp) * NN + n0 + nch * 8;
  const __bf16* bPtrB = bPtrA + NN;
  const int nb = nch * 8;

  int4 ra, rb0, rb1;
  ra  = *(const int4*)aPtr;
  rb0 = *(const int4*)bPtrA;
  rb1 = *(const int4*)bPtrB;

  auto storeB = [&](int buf, int4 xlo, int4 xhi) {
    const unsigned short* pa = (const unsigned short*)&xlo;
    const unsigned short* pb = (const unsigned short*)&xhi;
#pragma unroll
    for (int i = 0; i < 8; ++i) {
      const unsigned v = (unsigned)pa[i] | ((unsigned)pb[i] << 16);
      *(unsigned*)&lB[buf][(nb + i) * 40 + 2 * kp] = v;
    }
  };

  *(int4*)&lA[0][aLds] = ra;
  storeB(0, rb0, rb1);
  __syncthreads();

  const int lane = tid & 31, wv = tid >> 5;
  const int ml = lane & 15, hf = lane >> 4;
  v8f acc[4] = {};

  for (int kt = 0; kt < 11; ++kt) {
    const int cur = kt & 1, nxt = cur ^ 1;
    if (kt < 10) {
      const int ko = (kt + 1) * 32;
      ra  = *(const int4*)(aPtr + ko);
      rb0 = *(const int4*)(bPtrA + (size_t)ko * NN);
      rb1 = *(const int4*)(bPtrB + (size_t)ko * NN);
    }
    const __bf16* Ab = &lA[cur][0];
    const __bf16* Bb = &lB[cur][0];
    const v16bf bf = load_frag(Bb + (wv * 16 + ml) * 40, hf);
#pragma unroll
    for (int i = 0; i < 4; ++i) {
      const v16bf af = load_frag(Ab + (i * 16 + ml) * 40, hf);
      acc[i] = WMMA_BF16(af, bf, acc[i]);
    }
    if (kt < 10) {
      *(int4*)&lA[nxt][aLds] = ra;
      storeB(nxt, rb0, rb1);
    }
    __syncthreads();
  }

  const int n = n0 + wv * 16 + ml;
#pragma unroll
  for (int i = 0; i < 4; ++i) {
#pragma unroll
    for (int r = 0; r < 8; ++r) {
      const int m = m0 + i * 16 + hf * 8 + r;
      out[((size_t)b * M + m) * NN + n] = acc[i][r] + bias[m];
    }
  }
}

// ---------------- host driver ----------------
extern "C" void kernel_launch(void* const* d_in, const int* in_sizes, int n_in,
                              void* d_out, int out_size, void* d_ws, size_t ws_size,
                              hipStream_t stream) {
  const float* x    = (const float*)d_in[0];
  const unsigned char* mask = (const unsigned char*)d_in[1];  // jnp.bool_
  const float* adj  = (const float*)d_in[2];
  const float* Wr   = (const float*)d_in[3];
  const float* br   = (const float*)d_in[4];
  const float* Wu   = (const float*)d_in[5];
  const float* bu   = (const float*)d_in[6];
  const float* Wc   = (const float*)d_in[7];
  const float* bc   = (const float*)d_in[8];
  const float* Wout = (const float*)d_in[9];
  const float* bout = (const float*)d_in[10];

  float* outPred   = (float*)d_out;                       // [B,1,N,T]
  float* outStates = outPred + (size_t)BB * NN * TT;      // [1,B,H,N,T]

  char* p = (char*)d_ws;
  auto take = [&](size_t bytes) {
    char* r = p;
    p += (bytes + 255) & ~(size_t)255;
    return r;
  };
  __bf16* supF = (__bf16*)take((size_t)NN * NN * 2);
  __bf16* supB = (__bf16*)take((size_t)NN * NN * 2);
  __bf16* Wru  = (__bf16*)take((size_t)128 * CP * 2);
  __bf16* Wcb  = (__bf16*)take((size_t)64 * CP * 2);
  __bf16* zc   = (__bf16*)take((size_t)BB * CP * NN * 2);
  float*  h    = (float*)take((size_t)BB * HH * NN * 4);
  float*  ru   = (float*)take((size_t)BB * 128 * NN * 4);
  float*  cg   = (float*)take((size_t)BB * HH * NN * 4);
  float*  brbu = (float*)take(128 * 4);
  float*  invR = (float*)take(NN * 4);
  float*  invC = (float*)take(NN * 4);
  float*  pstg = (float*)take((size_t)16 * BB * NN * 4);
  float*  sstg = (float*)take((size_t)16 * BB * HH * NN * 4);

  k_sums<<<NN, 256, 0, stream>>>(adj, invR, invC);
  k_buildsup<<<(NN * NN) / 256, 256, 0, stream>>>(adj, invR, invC, supF, supB);
  k_buildw<<<(128 * CP + 255) / 256, 256, 0, stream>>>(Wr, Wu, Wc, br, bu, Wru, Wcb, brbu);
  k_zero<<<(BB * CP * NN + 255) / 256, 256, 0, stream>>>(h, zc);

  for (int t = 0; t < TT; ++t) {
    k_prep<<<(BB * NN) / 256, 256, 0, stream>>>(h, x, mask, Wout, bout, zc, pstg, sstg, t);
    // r/u path diffusion: order-1 (fwd & bwd merged), then order-2
    k_gemm_diff<<<dim3(8, 11, 2), 256, 0, stream>>>(zc, supF, supB, 0, C1, 0, 3 * C1);
    k_gemm_diff<<<dim3(8, 11, 2), 256, 0, stream>>>(zc, supF, supB, C1, 2 * C1, 3 * C1, 4 * C1);
    k_gemm_proj<<<dim3(8, 2, 8), 256, 0, stream>>>(Wru, zc, ru, brbu, 128);
    k_gate<<<(BB * HH * NN) / 256, 256, 0, stream>>>(ru, h, zc);
    // c path diffusion: [x_in, mask, r*h] (channels 0..1 reused)
    k_gemm_diff<<<dim3(8, 11, 2), 256, 0, stream>>>(zc, supF, supB, 0, C1, 0, 3 * C1);
    k_gemm_diff<<<dim3(8, 11, 2), 256, 0, stream>>>(zc, supF, supB, C1, 2 * C1, 3 * C1, 4 * C1);
    k_gemm_proj<<<dim3(8, 1, 8), 256, 0, stream>>>(Wcb, zc, cg, bc, 64);
    k_update<<<(BB * HH * NN) / 256, 256, 0, stream>>>(ru, cg, h);
    if ((t & 15) == 15) {
      k_flush_states<<<(BB * HH * NN) / 256, 256, 0, stream>>>(sstg, outStates, t - 15);
      k_flush_preds<<<(BB * NN) / 256, 256, 0, stream>>>(pstg, outPred, t - 15);
    }
  }
  (void)in_sizes; (void)n_in; (void)out_size; (void)ws_size;
}
